// Simpler2_37074157699278
// MI455X (gfx1250) — matrix operand order repaired and checked by reading, not compiled, not run on hardware
//
#include <hip/hip_runtime.h>
#include <hip/hip_bf16.h>
#include <math.h>

typedef __attribute__((ext_vector_type(16))) _Float16 v16h;
typedef __attribute__((ext_vector_type(8)))  float    v8f;
typedef __attribute__((ext_vector_type(4)))  float    f32x4;
typedef __attribute__((ext_vector_type(4)))  unsigned int u32x4;
typedef __attribute__((ext_vector_type(8)))  int      i32x8;
typedef __attribute__((ext_vector_type(4)))  int      i32x4;

union AF { v16h v; _Float16 e[16]; f32x4 q[2]; };
union CF { v8f  v; float    e[8];  };

__device__ inline v8f wmma16(v16h a, v16h b, v8f c) {
  // D(16x16 f32) = A(16x32 f16) * B(32x16 f16) + C
  return __builtin_amdgcn_wmma_f32_16x16x32_f16(false, a, false, b, (short)0, c, false, false);
}

// ---------------- fragment fillers (gfx1250 wave32 layouts, CDNA5 ISA §7.12.2) ----------------

// A operand: 16xK f16 row-major in LDS. Per-lane data = two contiguous 8-half runs
// (k = kb+h for h<8 ; k = 16+kb+(h-8) for h>=8) -> exactly 2x ds_load_b128.
__device__ inline v16h load_a_f16(const _Float16* __restrict__ A, int lda, int k0, int lane) {
  int m = lane & 15, kb = (lane & 16) ? 8 : 0;
  const _Float16* p = A + m * lda + k0 + kb;
  AF f;
  f.q[0] = *(const f32x4*)(p);
  f.q[1] = *(const f32x4*)(p + 16);
  return f.v;
}

// B operand from K-major (transposed) f16 source: B(k,n) = Bt[(n0+n)*ldk + k0+k].
// Per-lane data = two 8-half runs (k = kb+s, 16+kb+s-8), same as A -> 2x b128 loads. Works for LDS and global.
__device__ inline v16h load_bT(const _Float16* __restrict__ Bt, int ldk, int k0, int n0, int lane) {
  int n = lane & 15, kb = (lane & 16) ? 8 : 0;
  const _Float16* p = Bt + (size_t)(n0 + n) * ldk + k0 + kb;
  AF f;
  f.q[0] = *(const f32x4*)(p);
  f.q[1] = *(const f32x4*)(p + 16);
  return f.v;
}

// ---------------- Tensor Data Mover: DMA a flat f32 tile global->LDS ----------------
__device__ inline void tdm_load_f32_tile(float* lds_dst, const float* src, int n_elems) {
#if defined(__gfx1250__) && __has_builtin(__builtin_amdgcn_tensor_load_to_lds) && \
    __has_builtin(__builtin_amdgcn_s_wait_tensorcnt)
  unsigned long long ga = (unsigned long long)(uintptr_t)src;
  unsigned lds = (unsigned)(uintptr_t)lds_dst;          // LDS aperture: low 32 bits = byte offset
  unsigned td0 = (unsigned)n_elems;                     // <= 4096 elements (fits tile_dim0 u16)
  u32x4 g0;
  g0[0] = 1u;                                           // count=1, user descriptor
  g0[1] = lds;                                          // lds_addr (bytes)
  g0[2] = (unsigned)(ga & 0xffffffffu);                 // global_addr[31:0]
  g0[3] = (unsigned)((ga >> 32) & 0x01ffffffu) | (2u << 30); // global_addr[56:32] | type=2
  i32x8 g1;
  g1[0] = (int)(2u << 16);                              // workgroup_mask=0, data_size=4B
  g1[1] = (int)((td0 & 0xffffu) << 16);                 // tensor_dim0[15:0] (abar addr = 0)
  g1[2] = (int)(((td0 >> 16) & 0xffffu) | (1u << 16));  // tensor_dim0 hi | tensor_dim1=1
  g1[3] = (int)((td0 & 0xffffu) << 16);                 // tile_dim0 = n_elems
  g1[4] = 0;                                            // tile_dim1/2 unused (1D)
  g1[5] = (int)td0;                                     // tensor_dim0_stride lo
  g1[6] = 0;
  g1[7] = 0;
  i32x4 g2 = {0, 0, 0, 0}, g3 = {0, 0, 0, 0};
#if __clang_major__ >= 23
  i32x8 g4 = {0, 0, 0, 0, 0, 0, 0, 0};
  __builtin_amdgcn_tensor_load_to_lds(g0, g1, g2, g3, g4, 0);
#else
  __builtin_amdgcn_tensor_load_to_lds(g0, g1, g2, g3, 0);
#endif
  __builtin_amdgcn_s_wait_tensorcnt(0);
#else
  for (int i = threadIdx.x; i < n_elems; i += 32) lds_dst[i] = src[i];
#endif
}

// ---------------- weight prep: f32 [K][N] -> f16 transposed [N][K] ----------------
__global__ void cvtT_kernel(const float* __restrict__ W, int K, int N,
                            _Float16* __restrict__ WT, int total) {
  int idx = blockIdx.x * blockDim.x + threadIdx.x;
  if (idx >= total) return;
  int nn = idx / K, k = idx - nn * K;
  WT[(size_t)nn * K + k] = (_Float16)W[(size_t)k * N + nn];
}

// ---------------- embed + transpose: x[B,3,N] -> coords[B,N,3], feats[B,N,32] ----------------
__global__ void embed_kernel(const float* __restrict__ x, const float* __restrict__ ew,
                             const float* __restrict__ eb, float* __restrict__ coords,
                             float* __restrict__ feats, int B, int N) {
  int idx = blockIdx.x * blockDim.x + threadIdx.x;
  if (idx >= B * N) return;
  int b = idx / N, n = idx - b * N;
  float p0 = x[((size_t)b * 3 + 0) * N + n];
  float p1 = x[((size_t)b * 3 + 1) * N + n];
  float p2 = x[((size_t)b * 3 + 2) * N + n];
  coords[(size_t)idx * 3 + 0] = p0;
  coords[(size_t)idx * 3 + 1] = p1;
  coords[(size_t)idx * 3 + 2] = p2;
#pragma unroll
  for (int c = 0; c < 32; ++c)
    feats[(size_t)idx * 32 + c] = p0 * ew[c] + p1 * ew[32 + c] + p2 * ew[64 + c] + eb[c];
}

// ---------------- furthest point sampling: sequential argmax, dists resident in LDS ----------------
__global__ void fps_kernel(const float* __restrict__ coords, int N, int nsamp, int* __restrict__ fidx) {
  const int T = 256;
  int b = blockIdx.x, t = threadIdx.x;
  __shared__ float dists[8192];
  __shared__ float rv[T];
  __shared__ int   ri[T];
  __shared__ int   s_last;
  const float* C = coords + (size_t)b * N * 3;
  for (int i = t; i < N; i += T) dists[i] = 1e10f;
  if (t == 0) { fidx[b * nsamp] = 0; s_last = 0; }
  __syncthreads();
  for (int s = 1; s < nsamp; ++s) {
    int last = s_last;
    float lx = C[last * 3], ly = C[last * 3 + 1], lz = C[last * 3 + 2];
    float bv = -1.0f; int bi = 0x7fffffff;
    for (int i = t; i < N; i += T) {
      float dx = C[i * 3] - lx, dy = C[i * 3 + 1] - ly, dz = C[i * 3 + 2] - lz;
      float nd = fminf(dists[i], dx * dx + dy * dy + dz * dz);
      dists[i] = nd;
      if (nd > bv) { bv = nd; bi = i; }
    }
    rv[t] = bv; ri[t] = bi;
    __syncthreads();
    for (int off = T / 2; off > 0; off >>= 1) {
      if (t < off) {
        if (rv[t + off] > rv[t] || (rv[t + off] == rv[t] && ri[t + off] < ri[t])) {
          rv[t] = rv[t + off]; ri[t] = ri[t + off];
        }
      }
      __syncthreads();
    }
    if (t == 0) { s_last = ri[0]; fidx[b * nsamp + s] = ri[0]; }
    __syncthreads();
  }
}

// ---------------- KNN: 16 smallest distances per sampled center ----------------
__global__ void knn_kernel(const float* __restrict__ coords, int N, const int* __restrict__ fidx,
                           int npts, int* __restrict__ kidx, int total) {
  int q = blockIdx.x * blockDim.x + threadIdx.x;
  if (q >= total) return;
  int b = q / npts;
  const float* C = coords + (size_t)b * N * 3;
  int ci = fidx[q];
  float cx = C[ci * 3], cy = C[ci * 3 + 1], cz = C[ci * 3 + 2];
  float bd[16]; int bi[16];
#pragma unroll
  for (int k = 0; k < 16; ++k) { bd[k] = 1e30f; bi[k] = 0; }
  for (int j = 0; j < N; ++j) {
    float dx = C[j * 3] - cx, dy = C[j * 3 + 1] - cy, dz = C[j * 3 + 2] - cz;
    float d = dx * dx + dy * dy + dz * dz;
    if (d < bd[15]) {
      int p = 15;
      while (p > 0 && bd[p - 1] > d) { bd[p] = bd[p - 1]; bi[p] = bi[p - 1]; --p; }
      bd[p] = d; bi[p] = j;
    }
  }
#pragma unroll
  for (int k = 0; k < 16; ++k) kidx[(size_t)q * 16 + k] = bi[k];
}

// ---------------- gather neighborhoods ----------------
__global__ void gather_kernel(const float* __restrict__ coords, const float* __restrict__ feats,
                              int N, int C, const int* __restrict__ kidx, int npts,
                              float* __restrict__ gc, float* __restrict__ gf, int total) {
  int idx = blockIdx.x * blockDim.x + threadIdx.x;
  if (idx >= total) return;
  int g = idx >> 4;
  int b = g / npts;
  int src = kidx[idx];
  const float* cc = coords + ((size_t)b * N + src) * 3;
  gc[(size_t)idx * 3 + 0] = cc[0];
  gc[(size_t)idx * 3 + 1] = cc[1];
  gc[(size_t)idx * 3 + 2] = cc[2];
  const float* F = feats + ((size_t)b * N + src) * C;
  float* O = gf + (size_t)idx * C;
  for (int c = 0; c < C; ++c) O[c] = F[c];
}

// ---------------- transformer block: one wave32 per 16-point group ----------------
__device__ inline void layer_norm16(const float* __restrict__ x, _Float16* __restrict__ out, int dim,
                                    const float* __restrict__ s, const float* __restrict__ bb, int lane) {
  if (lane < 16) {
    const float* row = x + lane * dim;
    float m = 0.f;
    for (int c = 0; c < dim; ++c) m += row[c];
    m /= (float)dim;
    float v = 0.f;
    for (int c = 0; c < dim; ++c) { float d = row[c] - m; v += d * d; }
    v /= (float)dim;
    float inv = rsqrtf(v + 1e-5f);
    for (int c = 0; c < dim; ++c)
      out[lane * dim + c] = (_Float16)((row[c] - m) * inv * s[c] + bb[c]);
  }
}

__global__ void tfblock_kernel(float* __restrict__ gfeat, int dim, int heads, int dim_head, int inner,
                               const float* __restrict__ ln1_s, const float* __restrict__ ln1_b,
                               const float* __restrict__ ln2_s, const float* __restrict__ ln2_b,
                               const _Float16* __restrict__ qkvT,  // [3*inner][dim]  K-major
                               const _Float16* __restrict__ outT,  // [dim][inner]
                               const float* __restrict__ out_b,
                               const _Float16* __restrict__ ff1T,  // [dim][dim]
                               const float* __restrict__ ff1_b,
                               const _Float16* __restrict__ ff2T,  // [dim][dim]
                               const float* __restrict__ ff2_b) {
  const int lane = threadIdx.x;
  float* gx = gfeat + (size_t)blockIdx.x * 16 * dim;

  extern __shared__ char smem_raw[];
  float*    xbuf = (float*)smem_raw;          // 16*dim  f32  residual / att
  float*    smbf = xbuf + 16 * dim;           // 16*16   f32  attention scores
  _Float16* xn   = (_Float16*)(smbf + 256);   // 16*dim  f16  layernormed A tile
  _Float16* ob   = xn + 16 * dim;             // 16*inner f16 attention out / ffn hidden
  _Float16* qld  = ob + 16 * inner;           // 16*32   q tile (cols >= dh zero)
  _Float16* kld  = qld + 16 * 32;             // 16*32   k tile row-major == K-major for QK^T
  _Float16* vldT = kld + 16 * 32;             // 32*32   V transposed [d][j], j>=16 zero
  _Float16* pld  = vldT + 32 * 32;            // 16*32   softmax probs (cols >= 16 zero)

  const float scale = rsqrtf((float)dim_head);
  const int n = lane & 15, mo = (lane & 16) ? 8 : 0;

  // async tensor DMA of the group tile into LDS (TDM); pads zeroed in parallel
  tdm_load_f32_tile(xbuf, gx, 16 * dim);
  for (int i = lane; i < 16 * 32; i += 32) { qld[i] = (_Float16)0.f; kld[i] = (_Float16)0.f; pld[i] = (_Float16)0.f; }
  for (int i = lane; i < 32 * 32; i += 32) vldT[i] = (_Float16)0.f;
  __syncthreads();

  layer_norm16(xbuf, xn, dim, ln1_s, ln1_b, lane);
  __syncthreads();

  for (int h = 0; h < heads; ++h) {
    int qc = h * dim_head, kc = inner + h * dim_head, vc = 2 * inner + h * dim_head;
    for (int t = 0; t < dim_head; t += 16) {
      v8f cq = {}, ck = {}, cv = {};
      for (int k0 = 0; k0 < dim; k0 += 32) {
        v16h a = load_a_f16(xn, dim, k0, lane);
        cq = wmma16(a, load_bT(qkvT, dim, k0, qc + t, lane), cq);
        ck = wmma16(a, load_bT(qkvT, dim, k0, kc + t, lane), ck);
        cv = wmma16(a, load_bT(qkvT, dim, k0, vc + t, lane), cv);
      }
      CF uq, uk, uv; uq.v = cq; uk.v = ck; uv.v = cv;
#pragma unroll
      for (int r = 0; r < 8; ++r) {
        int m = r + mo;
        qld[m * 32 + t + n]  = (_Float16)uq.e[r];
        kld[m * 32 + t + n]  = (_Float16)uk.e[r];
        vldT[(t + n) * 32 + m] = (_Float16)uv.e[r];   // transposed store [d][j]
      }
    }
    __syncthreads();
    { // S = q @ k^T  (contraction over dim_head, zero-padded to 32)
      v8f cs = {};
      cs = wmma16(load_a_f16(qld, 32, 0, lane), load_bT(kld, 32, 0, 0, lane), cs);
      CF u; u.v = cs;
#pragma unroll
      for (int r = 0; r < 8; ++r) smbf[(r + mo) * 16 + n] = u.e[r] * scale;
    }
    __syncthreads();
    if (lane < 16) { // row softmax
      float mx = -1e30f;
#pragma unroll
      for (int j = 0; j < 16; ++j) mx = fmaxf(mx, smbf[lane * 16 + j]);
      float ex[16], s = 0.f;
#pragma unroll
      for (int j = 0; j < 16; ++j) { ex[j] = expf(smbf[lane * 16 + j] - mx); s += ex[j]; }
      float inv = 1.f / s;
#pragma unroll
      for (int j = 0; j < 16; ++j) pld[lane * 32 + j] = (_Float16)(ex[j] * inv);
    }
    __syncthreads();
    // O_h = P @ V  (contraction over 16 keys, zero-padded to 32; V stored d-major)
    for (int t = 0; t < dim_head; t += 16) {
      v8f co = {};
      co = wmma16(load_a_f16(pld, 32, 0, lane), load_bT(vldT, 32, 0, t, lane), co);
      CF u; u.v = co;
#pragma unroll
      for (int r = 0; r < 8; ++r) ob[(r + mo) * inner + h * dim_head + t + n] = (_Float16)u.e[r];
    }
    __syncthreads();
  }

  // att = o @ out_w + out_b + x  ->  xbuf
  for (int n0 = 0; n0 < dim; n0 += 16) {
    v8f c = {};
    for (int k0 = 0; k0 < inner; k0 += 32)
      c = wmma16(load_a_f16(ob, inner, k0, lane), load_bT(outT, inner, k0, n0, lane), c);
    CF u; u.v = c;
#pragma unroll
    for (int r = 0; r < 8; ++r) {
      int m = r + mo, col = n0 + n;
      xbuf[m * dim + col] += u.e[r] + out_b[col];
    }
  }
  __syncthreads();

  layer_norm16(xbuf, xn, dim, ln2_s, ln2_b, lane);
  __syncthreads();

  // y1 = gelu(xn @ ff1_w + ff1_b)  ->  ob (16 x dim, f16)
  for (int n0 = 0; n0 < dim; n0 += 16) {
    v8f c = {};
    for (int k0 = 0; k0 < dim; k0 += 32)
      c = wmma16(load_a_f16(xn, dim, k0, lane), load_bT(ff1T, dim, k0, n0, lane), c);
    CF u; u.v = c;
#pragma unroll
    for (int r = 0; r < 8; ++r) {
      int m = r + mo, col = n0 + n;
      float v = u.e[r] + ff1_b[col];
      ob[m * dim + col] = (_Float16)(0.5f * v * (1.f + erff(v * 0.70710678f)));
    }
  }
  __syncthreads();

  // out = y1 @ ff2_w + ff2_b + att  ->  global
  for (int n0 = 0; n0 < dim; n0 += 16) {
    v8f c = {};
    for (int k0 = 0; k0 < dim; k0 += 32)
      c = wmma16(load_a_f16(ob, dim, k0, lane), load_bT(ff2T, dim, k0, n0, lane), c);
    CF u; u.v = c;
#pragma unroll
    for (int r = 0; r < 8; ++r) {
      int m = r + mo, col = n0 + n;
      gx[m * dim + col] = u.e[r] + ff2_b[col] + xbuf[m * dim + col];
    }
  }
}

// ---------------- downsample (row 0 of each group) ----------------
__global__ void down_kernel(const float* __restrict__ gf, const float* __restrict__ gc, int dim,
                            const float* __restrict__ dw, const float* __restrict__ db,
                            float* __restrict__ pf, float* __restrict__ pc) {
  int g = blockIdx.x, t = threadIdx.x, od = 2 * dim;
  const float* in = gf + (size_t)g * 16 * dim; // neighbor 0 == center
  for (int c = t; c < od; c += blockDim.x) {
    float acc = db[c];
    for (int e = 0; e < dim; ++e) acc += in[e] * dw[(size_t)e * od + c];
    pf[(size_t)g * od + c] = fmaxf(acc, 0.f);
  }
  if (t < 3) pc[(size_t)g * 3 + t] = gc[(size_t)g * 16 * 3 + t];
}

// ---------------- mean-pool + classifier ----------------
__global__ void head_kernel(const float* __restrict__ pf, const float* __restrict__ cw,
                            const float* __restrict__ cb, float* __restrict__ out) {
  int b = blockIdx.x, t = threadIdx.x; // 512 threads
  __shared__ float mean[512];
  float s = 0.f;
#pragma unroll
  for (int p = 0; p < 16; ++p) s += pf[((size_t)b * 16 + p) * 512 + t];
  mean[t] = s * (1.f / 16.f);
  __syncthreads();
  if (t < 40) {
    float acc = cb[t];
    for (int c = 0; c < 512; ++c) acc += mean[c] * cw[c * 40 + t];
    out[b * 40 + t] = acc;
  }
}

// =====================================================================================

extern "C" void kernel_launch(void* const* d_in, const int* in_sizes, int n_in,
                              void* d_out, int out_size, void* d_ws, size_t ws_size,
                              hipStream_t stream) {
  // JAX pytree leaf order (dict keys sorted at every level; 'params' < 'x'):
  // 0:cls_b 1:cls_w 2:embed_b 3:embed_w
  // per stage: per block {ff1_b,ff1_w,ff2_b,ff2_w,ln1_b,ln1_s,ln2_b,ln2_s,out_b,out_w,qkv_w}, then down_b, down_w
  // last: x
  const float* cls_b   = (const float*)d_in[0];
  const float* cls_w   = (const float*)d_in[1];
  const float* embed_b = (const float*)d_in[2];
  const float* embed_w = (const float*)d_in[3];
  struct BlockP { const float *ff1_b,*ff1_w,*ff2_b,*ff2_w,*ln1_b,*ln1_s,*ln2_b,*ln2_s,*out_b,*out_w,*qkv_w; };
  const int bcount[4] = {1, 2, 1, 1};
  BlockP blocks[5];
  const float* down_b[4]; const float* down_w[4];
  int li = 4, bi = 0;
  for (int s = 0; s < 4; ++s) {
    for (int k = 0; k < bcount[s]; ++k) {
      BlockP& p = blocks[bi++];
      p.ff1_b = (const float*)d_in[li++]; p.ff1_w = (const float*)d_in[li++];
      p.ff2_b = (const float*)d_in[li++]; p.ff2_w = (const float*)d_in[li++];
      p.ln1_b = (const float*)d_in[li++]; p.ln1_s = (const float*)d_in[li++];
      p.ln2_b = (const float*)d_in[li++]; p.ln2_s = (const float*)d_in[li++];
      p.out_b = (const float*)d_in[li++]; p.out_w = (const float*)d_in[li++];
      p.qkv_w = (const float*)d_in[li++];
    }
    down_b[s] = (const float*)d_in[li++];
    down_w[s] = (const float*)d_in[li++];
  }
  const float* x = (const float*)d_in[li];

  // workspace partition (~31 MB)
  char* w = (char*)d_ws;
  float* pts_coord = (float*)w; w += (size_t)8 * 8192 * 3 * 4;
  float* pts_feat  = (float*)w; w += (size_t)8 * 8192 * 32 * 4;
  int*   fps_idx   = (int*)w;   w += (size_t)8 * 1024 * 4;
  int*   knn_idx   = (int*)w;   w += (size_t)8 * 1024 * 16 * 4;
  float* g_coords  = (float*)w; w += (size_t)8 * 1024 * 16 * 3 * 4;
  float* g_feats   = (float*)w; w += (size_t)8 * 1024 * 16 * 32 * 4;
  _Float16* wtpool = (_Float16*)w;

  const int B = 8;
  const int Ns[5]    = {8192, 1024, 256, 64, 16};
  const int dims[4]  = {32, 64, 128, 256};
  const int headsA[4]= {8, 16, 16, 32};
  const int dhA[4]   = {16, 16, 32, 32};

  // ---- pre-convert weights to f16 transposed (K-major) once ----
  struct BlockW { _Float16 *qkvT, *outT, *ff1T, *ff2T; };
  BlockW bw[5];
  {
    size_t off = 0; int bix = 0;
    for (int s = 0; s < 4; ++s) {
      int dim = dims[s], inner = headsA[s] * dhA[s];
      for (int k = 0; k < bcount[s]; ++k) {
        const BlockP& p = blocks[bix];
        BlockW& q = bw[bix];
        int tq = dim * 3 * inner, to = inner * dim, tf = dim * dim;
        q.qkvT = wtpool + off; off += tq;
        q.outT = wtpool + off; off += to;
        q.ff1T = wtpool + off; off += tf;
        q.ff2T = wtpool + off; off += tf;
        cvtT_kernel<<<(tq + 255) / 256, 256, 0, stream>>>(p.qkv_w, dim,   3 * inner, q.qkvT, tq);
        cvtT_kernel<<<(to + 255) / 256, 256, 0, stream>>>(p.out_w, inner, dim,       q.outT, to);
        cvtT_kernel<<<(tf + 255) / 256, 256, 0, stream>>>(p.ff1_w, dim,   dim,       q.ff1T, tf);
        cvtT_kernel<<<(tf + 255) / 256, 256, 0, stream>>>(p.ff2_w, dim,   dim,       q.ff2T, tf);
        ++bix;
      }
    }
  }

  embed_kernel<<<(B * 8192 + 255) / 256, 256, 0, stream>>>(x, embed_w, embed_b, pts_coord, pts_feat, B, 8192);

  int binx = 0;
  for (int s = 0; s < 4; ++s) {
    int N = Ns[s], npts = Ns[s + 1], dim = dims[s];
    int heads = headsA[s], dh = dhA[s], inner = heads * dh;

    fps_kernel<<<B, 256, 0, stream>>>(pts_coord, N, npts, fps_idx);
    knn_kernel<<<(B * npts + 127) / 128, 128, 0, stream>>>(pts_coord, N, fps_idx, npts, knn_idx, B * npts);
    gather_kernel<<<(B * npts * 16 + 255) / 256, 256, 0, stream>>>(pts_coord, pts_feat, N, dim, knn_idx,
                                                                   npts, g_coords, g_feats, B * npts * 16);

    size_t lds = (size_t)4 * (16 * dim + 256) + (size_t)2 * (16 * dim + 16 * inner + 16 * 32 * 3 + 32 * 32);
    for (int k = 0; k < bcount[s]; ++k) {
      const BlockP& p = blocks[binx + k];
      const BlockW& q = bw[binx + k];
      tfblock_kernel<<<B * npts, 32, lds, stream>>>(g_feats, dim, heads, dh, inner,
                                                    p.ln1_s, p.ln1_b, p.ln2_s, p.ln2_b,
                                                    q.qkvT, q.outT, p.out_b,
                                                    q.ff1T, p.ff1_b, q.ff2T, p.ff2_b);
    }
    binx += bcount[s];

    down_kernel<<<B * npts, 128, 0, stream>>>(g_feats, g_coords, dim, down_w[s], down_b[s], pts_feat, pts_coord);
  }

  head_kernel<<<B, 512, 0, stream>>>(pts_feat, cls_w, cls_b, (float*)d_out);
}
